// HierarchicalRouter_50534585205488
// MI455X (gfx1250) — compile-verified, matrix operand order checked by reading, not run on hardware
//
#include <hip/hip_runtime.h>
#include <hip/hip_bf16.h>
#include <math.h>

typedef __bf16 bf16;
typedef __attribute__((ext_vector_type(16))) __bf16 v16bf;
typedef __attribute__((ext_vector_type(8)))  __bf16 v8bf;
typedef __attribute__((ext_vector_type(8)))  float  v8f;

#define T_TOK 16384
#define H_DIM 4096
#define G_NUM 8
#define E_NUM 128
#define NCOL  144   // 128 expert cols + 8 group cols + 8 zero pad

// d_out flat layout (reference return order)
#define TOPK_OFF ((size_t)T_TOK * E_NUM)            // 2097152
#define EW_OFF   (TOPK_OFF + (size_t)T_TOK * 2)     // 2129920
#define VAR_OFF  (EW_OFF + (size_t)T_TOK * 2)       // 2162688
#define ENT_OFF  (VAR_OFF + 1)                      // 2162689

// ---------------------------------------------------------------- prep
__global__ void prep_weights(const float* __restrict__ Wg,
                             const float* __restrict__ We,
                             bf16* __restrict__ Wc) {
    size_t i = (size_t)blockIdx.x * blockDim.x + threadIdx.x;  // over 144*4096
    int r = (int)(i / H_DIM);
    int k = (int)(i % H_DIM);
    float v;
    if (r < E_NUM)              v = We[(size_t)r * H_DIM + k];
    else if (r < E_NUM + G_NUM) v = Wg[(size_t)(r - E_NUM) * H_DIM + k];
    else                        v = 0.0f;
    Wc[i] = (bf16)v;
}

__global__ void zero_stats(float* __restrict__ loadacc, float* __restrict__ entacc) {
    int i = threadIdx.x;
    if (i < E_NUM) loadacc[i] = 0.0f;
    if (i == 0)    entacc[0]  = 0.0f;
}

// ---------------------------------------------------------------- GEMM (WMMA bf16)
// A fragment per ISA 16-bit 16x32 layout: lane holds row (lane&15), K-half (lane>>4);
// elems 0..7 = K kb+h*8.., elems 8..15 = K kb+16+h*8..  (two contiguous 16B runs)
__device__ __forceinline__ v16bf load_a_frag(const float* __restrict__ p,
                                             int kb, int half) {
    const float4* q0 = (const float4*)(p + kb + half * 8);
    const float4* q1 = (const float4*)(p + kb + 16 + half * 8);
    float4 a0 = q0[0], a1 = q0[1], a2 = q1[0], a3 = q1[1];
    v16bf r;
    r[0]  = (bf16)a0.x; r[1]  = (bf16)a0.y; r[2]  = (bf16)a0.z; r[3]  = (bf16)a0.w;
    r[4]  = (bf16)a1.x; r[5]  = (bf16)a1.y; r[6]  = (bf16)a1.z; r[7]  = (bf16)a1.w;
    r[8]  = (bf16)a2.x; r[9]  = (bf16)a2.y; r[10] = (bf16)a2.z; r[11] = (bf16)a2.w;
    r[12] = (bf16)a3.x; r[13] = (bf16)a3.y; r[14] = (bf16)a3.z; r[15] = (bf16)a3.w;
    return r;
}

// Single-wave workgroups (no LDS / no barriers needed) so the scheduler can
// spread work across every WGP. Each wave: M=32 (two A fragments), N=144
// (9 subtiles) -> 18 WMMA per k-step; each B fragment feeds two WMMAs.
__global__ void __launch_bounds__(32)
gemm_kernel(const float* __restrict__ A, const bf16* __restrict__ Wc,
            float* __restrict__ Lg) {
    const int lane = threadIdx.x & 31;
    const int half = lane >> 4;
    const int l16  = lane & 15;

    const int rowbase = blockIdx.x * 32;
    const float* __restrict__ ar0 = A + (size_t)(rowbase + l16) * H_DIM;
    const float* __restrict__ ar1 = A + (size_t)(rowbase + 16 + l16) * H_DIM;

    v8f acc[18] = {};

    for (int kb = 0; kb < H_DIM; kb += 32) {
        if (kb + 256 < H_DIM) {   // ~8 k-steps (1KB) ahead on the HBM A stream
            __builtin_prefetch(ar0 + kb + 256, 0, 1);
            __builtin_prefetch(ar1 + kb + 256, 0, 1);
        }
        v16bf af0 = load_a_frag(ar0, kb, half);
        v16bf af1 = load_a_frag(ar1, kb, half);

        #pragma unroll
        for (int n = 0; n < 9; ++n) {
            const int col = n * 16 + l16;
            const bf16* __restrict__ wr = Wc + (size_t)col * H_DIM;
            v8bf b0 = *(const v8bf*)(wr + kb + half * 8);
            v8bf b1 = *(const v8bf*)(wr + kb + 16 + half * 8);
            v16bf bfrag = __builtin_shufflevector(b0, b1,
                0, 1, 2, 3, 4, 5, 6, 7, 8, 9, 10, 11, 12, 13, 14, 15);
            acc[2 * n]     = __builtin_amdgcn_wmma_f32_16x16x32_bf16(
                false, af0, false, bfrag, (short)0, acc[2 * n],     false, false);
            acc[2 * n + 1] = __builtin_amdgcn_wmma_f32_16x16x32_bf16(
                false, af1, false, bfrag, (short)0, acc[2 * n + 1], false, false);
        }
    }

    // C layout: lanes 0-15 -> M=j, lanes 16-31 -> M=8+j; N = lane&15
    #pragma unroll
    for (int n = 0; n < 9; ++n) {
        #pragma unroll
        for (int f = 0; f < 2; ++f) {
            const int orow_base = rowbase + f * 16 + half * 8;
            #pragma unroll
            for (int j = 0; j < 8; ++j) {
                Lg[(size_t)(orow_base + j) * NCOL + (n * 16 + l16)] =
                    acc[2 * n + f][j];
            }
        }
    }
}

// ---------------------------------------------------------------- routing helpers
__device__ __forceinline__ bool pref_(float a, int ai, float b, int bi) {
    return (a > b) || (a == b && ai < bi);
}
__device__ __forceinline__ void merge2(float& v1, int& i1, float& v2, int& i2,
                                       float ov1, int oi1, float ov2, int oi2) {
    float m1v, m2v; int m1i, m2i;
    if (pref_(v1, i1, ov1, oi1)) {
        m1v = v1; m1i = i1;
        if (pref_(v2, i2, ov1, oi1)) { m2v = v2;  m2i = i2;  }
        else                         { m2v = ov1; m2i = oi1; }
    } else {
        m1v = ov1; m1i = oi1;
        if (pref_(ov2, oi2, v1, i1)) { m2v = ov2; m2i = oi2; }
        else                         { m2v = v1;  m2i = i1;  }
    }
    v1 = m1v; i1 = m1i; v2 = m2v; i2 = m2i;
}
__device__ __forceinline__ void insert2(float& v1, int& i1, float& v2, int& i2,
                                        float v, int i) {
    if (pref_(v, i, v1, i1)) { v2 = v1; i2 = i1; v1 = v; i1 = i; }
    else if (pref_(v, i, v2, i2)) { v2 = v; i2 = i; }
}

// ---------------------------------------------------------------- routing (wave per token)
__global__ void __launch_bounds__(256)
route_kernel(const float* __restrict__ Lg, float* __restrict__ out,
             float* __restrict__ loadacc, float* __restrict__ entacc) {
    __shared__ float s_load[E_NUM];
    __shared__ float s_ent;
    const int tid = threadIdx.x;
    if (tid < E_NUM) s_load[tid] = 0.0f;
    if (tid == 0)    s_ent = 0.0f;
    __syncthreads();

    const int wave = tid >> 5, lane = tid & 31;
    const int t = blockIdx.x * 8 + wave;
    const float NEG = -INFINITY;
    const float* __restrict__ row = Lg + (size_t)t * NCOL;

    // ---- group top-2 (8 group logits live in cols 128..135) ----
    float gv1 = (lane < G_NUM) ? row[E_NUM + lane] : NEG;
    int   gi1 = lane;
    float gv2 = NEG; int gi2 = 0x7fffffff;
    #pragma unroll
    for (int m = 16; m; m >>= 1) {
        float ov1 = __shfl_xor(gv1, m, 32); int oi1 = __shfl_xor(gi1, m, 32);
        float ov2 = __shfl_xor(gv2, m, 32); int oi2 = __shfl_xor(gi2, m, 32);
        merge2(gv1, gi1, gv2, gi2, ov1, oi1, ov2, oi2);
    }
    const float ge  = __expf(gv2 - gv1);
    const float gw1 = 1.0f / (1.0f + ge);
    const float gw2 = ge   / (1.0f + ge);
    const float lw1 = __logf(gw1 + 1e-8f);
    const float lw2 = __logf(gw2 + 1e-8f);

    // ---- adjusted expert logits: 4 experts per lane ----
    float4 x4 = *(const float4*)(row + lane * 4);
    float adj[4];
    {
        float xs[4] = {x4.x, x4.y, x4.z, x4.w};
        #pragma unroll
        for (int j = 0; j < 4; ++j) {
            const int e = lane * 4 + j;
            const int g = e >> 4;
            adj[j] = (g == gi1) ? xs[j] + lw1 : ((g == gi2) ? xs[j] + lw2 : NEG);
        }
    }
    *(float4*)(out + (size_t)t * E_NUM + lane * 4) =
        make_float4(adj[0], adj[1], adj[2], adj[3]);

    // ---- expert top-2 ----
    float v1 = adj[0]; int i1 = lane * 4;
    float v2 = NEG;    int i2 = 0x7fffffff;
    insert2(v1, i1, v2, i2, adj[1], lane * 4 + 1);
    insert2(v1, i1, v2, i2, adj[2], lane * 4 + 2);
    insert2(v1, i1, v2, i2, adj[3], lane * 4 + 3);
    #pragma unroll
    for (int m = 16; m; m >>= 1) {
        float ov1 = __shfl_xor(v1, m, 32); int oi1 = __shfl_xor(i1, m, 32);
        float ov2 = __shfl_xor(v2, m, 32); int oi2 = __shfl_xor(i2, m, 32);
        merge2(v1, i1, v2, i2, ov1, oi1, ov2, oi2);
    }
    const float te = __expf(v2 - v1);
    const float w1 = 1.0f / (1.0f + te);
    const float w2 = te   / (1.0f + te);
    if (lane == 0) {
        out[TOPK_OFF + (size_t)t * 2]     = (float)i1;
        out[TOPK_OFF + (size_t)t * 2 + 1] = (float)i2;
        out[EW_OFF   + (size_t)t * 2]     = w1;
        out[EW_OFF   + (size_t)t * 2 + 1] = w2;
    }

    // ---- full softmax over 128 for stats ----
    const float mmax = v1;  // global max (broadcast after butterfly)
    float pe[4]; float sl = 0.0f;
    #pragma unroll
    for (int j = 0; j < 4; ++j) { pe[j] = __expf(adj[j] - mmax); sl += pe[j]; }
    #pragma unroll
    for (int m = 16; m; m >>= 1) sl += __shfl_xor(sl, m, 32);
    const float inv = 1.0f / sl;
    float ent = 0.0f;
    #pragma unroll
    for (int j = 0; j < 4; ++j) {
        const float p = pe[j] * inv;
        ent -= p * __logf(p + 1e-8f);
        atomicAdd(&s_load[lane * 4 + j], p);
    }
    #pragma unroll
    for (int m = 16; m; m >>= 1) ent += __shfl_xor(ent, m, 32);
    if (lane == 0) atomicAdd(&s_ent, ent);

    __syncthreads();
    if (tid < E_NUM) atomicAdd(&loadacc[tid], s_load[tid]);
    if (tid == 0)    atomicAdd(entacc, s_ent);
}

// ---------------------------------------------------------------- finalize (var + entropy)
__global__ void finalize_kernel(const float* __restrict__ loadacc,
                                const float* __restrict__ entacc,
                                float* __restrict__ out) {
    const int lane = threadIdx.x;  // 32 threads
    float x[4]; float s = 0.0f;
    #pragma unroll
    for (int j = 0; j < 4; ++j) {
        x[j] = loadacc[lane * 4 + j] * (1.0f / T_TOK);
        s += x[j];
    }
    #pragma unroll
    for (int m = 16; m; m >>= 1) s += __shfl_xor(s, m, 32);
    const float mean = s * (1.0f / E_NUM);
    float d = 0.0f;
    #pragma unroll
    for (int j = 0; j < 4; ++j) { float t = x[j] - mean; d += t * t; }
    #pragma unroll
    for (int m = 16; m; m >>= 1) d += __shfl_xor(d, m, 32);
    if (lane == 0) {
        out[VAR_OFF] = d * (1.0f / (E_NUM - 1));   // unbiased (ddof=1)
        out[ENT_OFF] = entacc[0] * (1.0f / T_TOK);
    }
}

// ---------------------------------------------------------------- launch
extern "C" void kernel_launch(void* const* d_in, const int* in_sizes, int n_in,
                              void* d_out, int out_size, void* d_ws, size_t ws_size,
                              hipStream_t stream) {
    const float* hidden = (const float*)d_in[0];
    const float* Wg     = (const float*)d_in[1];
    const float* We     = (const float*)d_in[2];
    float* out = (float*)d_out;

    char* ws = (char*)d_ws;
    bf16*  Wc      = (bf16*)ws;                                        // 144*4096*2  = 1,179,648 B
    float* Lg      = (float*)(ws + (size_t)NCOL * H_DIM * 2);          // T*144*4     = 9,437,184 B
    float* loadacc = (float*)(ws + (size_t)NCOL * H_DIM * 2
                                 + (size_t)T_TOK * NCOL * 4);          // 128 floats
    float* entacc  = loadacc + E_NUM;                                  // 1 float

    prep_weights   <<<(NCOL * H_DIM) / 256, 256, 0, stream>>>(Wg, We, Wc);
    zero_stats     <<<1, 256, 0, stream>>>(loadacc, entacc);
    gemm_kernel    <<<T_TOK / 32, 32, 0, stream>>>(hidden, Wc, Lg);
    route_kernel   <<<T_TOK / 8, 256, 0, stream>>>(Lg, out, loadacc, entacc);
    finalize_kernel<<<1, 32, 0, stream>>>(loadacc, entacc, out);
}
